// Multi_head_attention_21646635171939
// MI455X (gfx1250) — matrix operand order repaired and checked by reading, not compiled, not run on hardware
//
#include <hip/hip_runtime.h>
#include <hip/hip_bf16.h>
#include <cstdint>

// ---------------------------------------------------------------------------
// Multi-head causal attention for MI455X (gfx1250, wave32, WMMA + TDM).
//   B=4, T=2048, D=1024, H=16, hd=64
// Pipeline:
//   1) gemm_qkv<float,0>: Qh = q@wq^T + bq   -> [B,H,T,64]  f16
//   2) gemm_qkv<float,0>: Kh = k@wk^T + bk   -> [B,H,T,64]  f16
//   3) gemm_qkv<float,1>: Vt = v@wv^T + bv   -> [B,H,64,T]  f16 (transposed)
//   4) flash_attn:        Oi = softmax(QK^T/8, causal) @ V -> [B,T,D] f16
//        K/V tiles staged into LDS by the Tensor Data Mover, double-buffered
//        on TENSORcnt (falls back to cooperative copies if builtin missing).
//   5) gemm_qkv<f16,2>:   out = Oi@wo^T + bo -> [B,T,D] f32 (d_out)
// All matrix math on v_wmma_f32_16x16x32_f16 (f32 accumulate).
// ---------------------------------------------------------------------------

typedef __attribute__((ext_vector_type(16))) _Float16     v16h;
typedef __attribute__((ext_vector_type(8)))  _Float16     v8h;
typedef __attribute__((ext_vector_type(4)))  _Float16     v4h;
typedef __attribute__((ext_vector_type(8)))  float        v8f;
typedef __attribute__((ext_vector_type(4)))  unsigned int v4u;
typedef __attribute__((ext_vector_type(8)))  int          v8i;
typedef __attribute__((ext_vector_type(4)))  int          v4i;

#define D_MODEL  1024
#define N_HEADS  16
#define HEAD_DIM 64
#define SEQ      2048
#define BATCH    4
#define MROWS    (BATCH * SEQ)   // 8192

// TDM availability: device compile on gfx1250 with the builtin present.
#if defined(__HIP_DEVICE_COMPILE__) && defined(__gfx1250__) && \
    __has_builtin(__builtin_amdgcn_tensor_load_to_lds)
#define HAVE_TDM 1
#else
#define HAVE_TDM 0
#endif

static __device__ __forceinline__ v16h cat8(v8h lo, v8h hi) {
  return __builtin_shufflevector(lo, hi, 0,1,2,3,4,5,6,7,8,9,10,11,12,13,14,15);
}

static __device__ __forceinline__ v8f wmma_f16(v16h a, v16h b, v8f c) {
  // (neg_a, A, neg_b, B, c_mod, C, reuse_a, reuse_b)
  return __builtin_amdgcn_wmma_f32_16x16x32_f16(false, a, false, b, (short)0, c,
                                                false, false);
}

static __device__ __forceinline__ float rmax16(float x) {
  x = fmaxf(x, __shfl_xor(x, 8, 32));
  x = fmaxf(x, __shfl_xor(x, 4, 32));
  x = fmaxf(x, __shfl_xor(x, 2, 32));
  x = fmaxf(x, __shfl_xor(x, 1, 32));
  return x;
}
static __device__ __forceinline__ float rsum16(float x) {
  x += __shfl_xor(x, 8, 32);
  x += __shfl_xor(x, 4, 32);
  x += __shfl_xor(x, 2, 32);
  x += __shfl_xor(x, 1, 32);
  return x;
}

#if HAVE_TDM
// Issue a 2-D TDM tile load: tile_d0 x tile_d1 elements of 2 bytes,
// row stride `stride0` (elements), from `gptr` (tile start) into LDS
// offset `lds_off`. Descriptor per CDNA5 ISA D# group0/group1 layout.
static __device__ __forceinline__ void tdm_load_2d(uint32_t lds_off,
                                                   const void* gptr,
                                                   uint32_t tensor_d0,
                                                   uint32_t tensor_d1,
                                                   uint64_t stride0,
                                                   uint32_t tile_d0,
                                                   uint32_t tile_d1) {
  const uint64_t ga = (uint64_t)(uintptr_t)gptr;
  v4u g0;
  g0[0] = 1u;                                              // count=1
  g0[1] = lds_off;                                         // lds_addr
  g0[2] = (uint32_t)ga;                                    // global_addr lo
  g0[3] = (uint32_t)((ga >> 32) & 0x01FFFFFFu) | (2u << 30);  // ga hi | type=2
  v8i g1;
  g1[0] = (int)(1u << 16);                                 // data_size=1 (2B)
  g1[1] = (int)((tensor_d0 & 0xFFFFu) << 16);              // tensor_dim0 lo16
  g1[2] = (int)((tensor_d0 >> 16) | ((tensor_d1 & 0xFFFFu) << 16));
  g1[3] = (int)((tensor_d1 >> 16) | (tile_d0 << 16));      // | tile_dim0
  g1[4] = (int)tile_d1;                                    // tile_dim1, dim2=0
  g1[5] = (int)(uint32_t)(stride0 & 0xFFFFFFFFu);          // stride0 lo32
  g1[6] = (int)(uint32_t)((stride0 >> 32) & 0xFFFFu);      // stride0 hi16
  g1[7] = 0;
  const v4i z4 = {0, 0, 0, 0};
#if defined(__clang_major__) && (__clang_major__ >= 23)
  const v8i z8 = {0, 0, 0, 0, 0, 0, 0, 0};
  __builtin_amdgcn_tensor_load_to_lds(g0, g1, z4, z4, z8, 0);
#else
  __builtin_amdgcn_tensor_load_to_lds(g0, g1, z4, z4, 0);
#endif
}
#endif  // HAVE_TDM

// ---------------------------------------------------------------------------
// GEMM: Y = X (M x K) @ W^T (K x N) + bias,  W stored row-major [N, K].
// Block tile 128x128, 8 waves (2 M x 4 N), wave tile 64x32 (4x2 fragments).
// MODE 0: write f16 [B,H,T,hd];  MODE 1: write f16 [B,H,hd,T];
// MODE 2: write f32 row-major [M,N].
// ---------------------------------------------------------------------------
template <typename TX, int MODE>
__global__ __launch_bounds__(256) void gemm_qkv(const TX* __restrict__ X,
                                                const float* __restrict__ W,
                                                const float* __restrict__ bias,
                                                void* __restrict__ Yout) {
  constexpr int K = D_MODEL;
  __shared__ _Float16 Xs[128][40];   // 32 data halfs + pad
  __shared__ _Float16 Ws[128][40];

  const int tid  = threadIdx.x;
  const int lane = tid & 31;
  const int wave = tid >> 5;
  const int rsel = lane & 15;        // row (A) / col (B,C,D) selector
  const int hsel = lane >> 4;        // half-wave selector
  const int wM = (wave >> 2) * 64;
  const int wN = (wave & 3) * 32;
  const int m0 = blockIdx.y * 128;
  const int n0 = blockIdx.x * 128;

  const v8f vzero = {0.f, 0.f, 0.f, 0.f, 0.f, 0.f, 0.f, 0.f};
  v8f acc[4][2];
#pragma unroll
  for (int mi = 0; mi < 4; ++mi)
#pragma unroll
    for (int ni = 0; ni < 2; ++ni) acc[mi][ni] = vzero;

  for (int k0 = 0; k0 < K; k0 += 32) {
    // ---- stage 128x32 tiles of X and W into LDS (fp32 -> f16 convert) ----
#pragma unroll
    for (int r = 0; r < 4; ++r) {
      const int i   = tid + r * 256;     // 0..1023
      const int row = i >> 3;            // 0..127
      const int c4  = (i & 7) << 2;      // 0,4,...,28
      const TX*    sx = X + (size_t)(m0 + row) * K + k0 + c4;
      const float* sw = W + (size_t)(n0 + row) * K + k0 + c4;
      v4h hx, hw;
      if constexpr (sizeof(TX) == 4) {
        const float4 fx = *(const float4*)sx;
        hx[0] = (_Float16)fx.x; hx[1] = (_Float16)fx.y;
        hx[2] = (_Float16)fx.z; hx[3] = (_Float16)fx.w;
      } else {
        hx = *(const v4h*)sx;
      }
      const float4 fw = *(const float4*)sw;
      hw[0] = (_Float16)fw.x; hw[1] = (_Float16)fw.y;
      hw[2] = (_Float16)fw.z; hw[3] = (_Float16)fw.w;
      *(v4h*)&Xs[row][c4] = hx;
      *(v4h*)&Ws[row][c4] = hw;
      if (k0 + 32 < K) {   // hint next K tile toward the caches
        __builtin_prefetch((const void*)(sx + 32), 0, 1);
        __builtin_prefetch((const void*)(sw + 32), 0, 1);
      }
    }
    __syncthreads();

    // ---- fragments from LDS ----
    // A (16x32 f16): lane<16 holds row rsel, K {kb..kb+7, kb+16..kb+23}, kb=hsel*8
    v16h a[4];
#pragma unroll
    for (int mi = 0; mi < 4; ++mi) {
      const _Float16* p = &Xs[wM + mi * 16 + rsel][hsel * 8];
      a[mi] = cat8(*(const v8h*)p, *(const v8h*)(p + 16));
    }
    // B (32x16 f16): lane holds col rsel, K {kb..kb+7, kb+16..kb+23}, kb=hsel*8
    v16h b[2];
#pragma unroll
    for (int ni = 0; ni < 2; ++ni) {
      const _Float16* p = &Ws[wN + ni * 16 + rsel][hsel * 8];
      b[ni] = cat8(*(const v8h*)p, *(const v8h*)(p + 16));
    }
#pragma unroll
    for (int mi = 0; mi < 4; ++mi)
#pragma unroll
      for (int ni = 0; ni < 2; ++ni)
        acc[mi][ni] = wmma_f16(a[mi], b[ni], acc[mi][ni]);
    __syncthreads();
  }

  // ---- epilogue: bias + layout remap ----
#pragma unroll
  for (int mi = 0; mi < 4; ++mi) {
#pragma unroll
    for (int ni = 0; ni < 2; ++ni) {
#pragma unroll
      for (int v = 0; v < 8; ++v) {
        const int m = m0 + wM + mi * 16 + hsel * 8 + v;
        const int n = n0 + wN + ni * 16 + rsel;
        const float y = acc[mi][ni][v] + bias[n];
        if constexpr (MODE == 0) {
          _Float16* Y = (_Float16*)Yout;
          const int bb = m >> 11, t = m & 2047, h = n >> 6, d = n & 63;
          Y[(((size_t)bb * N_HEADS + h) * SEQ + t) * HEAD_DIM + d] = (_Float16)y;
        } else if constexpr (MODE == 1) {
          _Float16* Y = (_Float16*)Yout;
          const int bb = m >> 11, t = m & 2047, h = n >> 6, d = n & 63;
          Y[(((size_t)bb * N_HEADS + h) * HEAD_DIM + d) * SEQ + t] = (_Float16)y;
        } else {
          ((float*)Yout)[(size_t)m * D_MODEL + n] = y;
        }
      }
    }
  }
}

// ---------------------------------------------------------------------------
// Causal flash attention. Grid (T/64, B*H); 4 waves/block, each wave owns
// 16 query rows. Q fragments live in registers for the whole key loop.
// K (32x64) and V^T (64x32) tiles are DMA'd into LDS by the Tensor Data
// Mover, double-buffered on TENSORcnt; one DMA per tile per block instead
// of 4 redundant per-wave global fetches. Online softmax in exp2 domain;
// P transposed C->A layout through LDS.
// ---------------------------------------------------------------------------
__global__ __launch_bounds__(128) void flash_attn(const _Float16* __restrict__ Qh,
                                                  const _Float16* __restrict__ Kh,
                                                  const _Float16* __restrict__ Vt,
                                                  _Float16* __restrict__ O) {
  __shared__ _Float16 Kbuf[2][32][64];   // [buf][key][hd]   8 KB
  __shared__ _Float16 Vbuf[2][64][32];   // [buf][hd][key]   8 KB
  __shared__ _Float16 Ps[4][16][40];     // per-wave P tile (padded)

  const int tid  = threadIdx.x;
  const int lane = tid & 31;
  const int wave = tid >> 5;
  const int rsel = lane & 15;
  const int hsel = lane >> 4;
  const int bh = blockIdx.y;             // b*16 + h
  const int q0 = blockIdx.x * 64;
  const int qb = q0 + wave * 16;

  const _Float16* Qp = Qh + (size_t)bh * SEQ * HEAD_DIM;
  const _Float16* Kp = Kh + (size_t)bh * SEQ * HEAD_DIM;
  const _Float16* Vp = Vt + (size_t)bh * HEAD_DIM * SEQ;

  // Q A-fragments for the two 32-wide head-dim chunks
  v16h qf[2];
#pragma unroll
  for (int c = 0; c < 2; ++c) {
    const _Float16* p = Qp + (size_t)(qb + rsel) * HEAD_DIM + c * 32 + hsel * 8;
    qf[c] = cat8(*(const v8h*)p, *(const v8h*)(p + 16));
  }

  const v8f vzero = {0.f, 0.f, 0.f, 0.f, 0.f, 0.f, 0.f, 0.f};
  v8f o[4] = {vzero, vzero, vzero, vzero};
  float mrow[8], lrow[8];
#pragma unroll
  for (int v = 0; v < 8; ++v) { mrow[v] = -3.0e38f; lrow[v] = 0.f; }

  const float sscale = 0.18033688011112042f;  // (1/sqrt(64)) * log2(e)
  const int nblk = (q0 + 64) / 32;            // causal bound for this block

#if HAVE_TDM
  if (wave == 0) {  // prologue: DMA block 0 into buffer 0
    tdm_load_2d((uint32_t)(uintptr_t)&Kbuf[0][0][0], Kp,
                HEAD_DIM, SEQ, HEAD_DIM, HEAD_DIM, 32);
    tdm_load_2d((uint32_t)(uintptr_t)&Vbuf[0][0][0], Vp,
                SEQ, HEAD_DIM, SEQ, 32, HEAD_DIM);
  }
#endif

  for (int i = 0; i < nblk; ++i) {
    const int k0   = i * 32;
    const int bsel = i & 1;
    __syncthreads();   // all waves done reading buf[(i+1)&1] from iter i-1
#if HAVE_TDM
    if (wave == 0) {
      if (i + 1 < nblk) {  // prefetch next block into the other buffer
        const int kn = (i + 1) * 32;
        tdm_load_2d((uint32_t)(uintptr_t)&Kbuf[bsel ^ 1][0][0],
                    Kp + (size_t)kn * HEAD_DIM,
                    HEAD_DIM, SEQ, HEAD_DIM, HEAD_DIM, 32);
        tdm_load_2d((uint32_t)(uintptr_t)&Vbuf[bsel ^ 1][0][0],
                    Vp + kn, SEQ, HEAD_DIM, SEQ, 32, HEAD_DIM);
        __builtin_amdgcn_s_wait_tensorcnt((short)2);  // block i complete
      } else {
        __builtin_amdgcn_s_wait_tensorcnt((short)0);
      }
    }
#else
    // Fallback: cooperative copy of block i tiles (K tile is contiguous).
    {
      const _Float16* ksrc = Kp + (size_t)k0 * HEAD_DIM;
      _Float16* kdst = &Kbuf[bsel][0][0];
      for (int j = tid; j < (32 * HEAD_DIM) / 8; j += 128)
        *(v8h*)(kdst + j * 8) = *(const v8h*)(ksrc + j * 8);
      for (int j = tid; j < (HEAD_DIM * 32) / 8; j += 128) {
        const int row = j >> 2, ch = (j & 3) * 8;
        *(v8h*)&Vbuf[bsel][row][ch] =
            *(const v8h*)(Vp + (size_t)row * SEQ + k0 + ch);
      }
    }
#endif
    __syncthreads();   // block i tiles visible to all waves

    // ---- S = Q @ K^T for a 16x32 score tile (two 16-col fragments) ----
    v8f s[2] = {vzero, vzero};
#pragma unroll
    for (int fi = 0; fi < 2; ++fi) {
#pragma unroll
      for (int c = 0; c < 2; ++c) {
        const _Float16* p = &Kbuf[bsel][fi * 16 + rsel][c * 32 + hsel * 8];
        const v16h kf = cat8(*(const v8h*)p, *(const v8h*)(p + 16));
        s[fi] = wmma_f16(qf[c], kf, s[fi]);
      }
    }

    // ---- causal mask + online softmax (per C-layout row) ----
#pragma unroll
    for (int v = 0; v < 8; ++v) {
      const int qrow = qb + hsel * 8 + v;
      const float x0 = (k0 + rsel      <= qrow) ? s[0][v] * sscale : -3.0e38f;
      const float x1 = (k0 + 16 + rsel <= qrow) ? s[1][v] * sscale : -3.0e38f;
      const float mnew  = fmaxf(mrow[v], rmax16(fmaxf(x0, x1)));
      const float alpha = exp2f(mrow[v] - mnew);
      const float p0 = exp2f(x0 - mnew);
      const float p1 = exp2f(x1 - mnew);
      lrow[v] = lrow[v] * alpha + rsum16(p0 + p1);
      mrow[v] = mnew;
#pragma unroll
      for (int ni = 0; ni < 4; ++ni) o[ni][v] *= alpha;
      s[0][v] = p0;
      s[1][v] = p1;
    }

    // ---- C-layout P -> LDS -> A-layout fragment ----
#pragma unroll
    for (int v = 0; v < 8; ++v) {
      Ps[wave][hsel * 8 + v][rsel]      = (_Float16)s[0][v];
      Ps[wave][hsel * 8 + v][16 + rsel] = (_Float16)s[1][v];
    }
    __syncthreads();   // uniform trip count across block
    const _Float16* pp = &Ps[wave][rsel][hsel * 8];
    const v16h pa = cat8(*(const v8h*)pp, *(const v8h*)(pp + 16));

    // ---- O += P @ V ----
#pragma unroll
    for (int ni = 0; ni < 4; ++ni) {
      const _Float16* vp = &Vbuf[bsel][ni * 16 + rsel][hsel * 8];
      const v16h vf = cat8(*(const v8h*)vp, *(const v8h*)(vp + 16));
      o[ni] = wmma_f16(pa, vf, o[ni]);
    }
  }

  // ---- normalize and write [B,T,D] f16 ----
  const int b = bh >> 4, h = bh & 15;
#pragma unroll
  for (int v = 0; v < 8; ++v) {
    const float inv = 1.0f / lrow[v];
    const int t = qb + hsel * 8 + v;
    _Float16* orow = O + ((size_t)b * SEQ + t) * D_MODEL + h * HEAD_DIM;
#pragma unroll
    for (int ni = 0; ni < 4; ++ni)
      orow[ni * 16 + rsel] = (_Float16)(o[ni][v] * inv);
  }
}

// ---------------------------------------------------------------------------
extern "C" void kernel_launch(void* const* d_in, const int* in_sizes, int n_in,
                              void* d_out, int out_size, void* d_ws,
                              size_t ws_size, hipStream_t stream) {
  (void)in_sizes; (void)n_in; (void)out_size; (void)ws_size;

  const float* q  = (const float*)d_in[0];
  const float* k  = (const float*)d_in[1];
  const float* v  = (const float*)d_in[2];
  const float* wq = (const float*)d_in[3];
  const float* bq = (const float*)d_in[4];
  const float* wk = (const float*)d_in[5];
  const float* bk = (const float*)d_in[6];
  const float* wv = (const float*)d_in[7];
  const float* bv = (const float*)d_in[8];
  const float* wo = (const float*)d_in[9];
  const float* bo = (const float*)d_in[10];

  const size_t NT = (size_t)BATCH * N_HEADS * SEQ * HEAD_DIM;  // 8.39M elems
  _Float16* Qh = (_Float16*)d_ws;
  _Float16* Kh = Qh + NT;
  _Float16* Vt = Kh + NT;
  _Float16* Oi = Vt + NT;   // total 64 MB of f16 workspace

  const dim3 gp(D_MODEL / 128, MROWS / 128);  // (8, 64)
  gemm_qkv<float, 0><<<gp, 256, 0, stream>>>(q, wq, bq, (void*)Qh);
  gemm_qkv<float, 0><<<gp, 256, 0, stream>>>(k, wk, bk, (void*)Kh);
  gemm_qkv<float, 1><<<gp, 256, 0, stream>>>(v, wv, bv, (void*)Vt);

  flash_attn<<<dim3(SEQ / 64, BATCH * N_HEADS), 128, 0, stream>>>(Qh, Kh, Vt, Oi);

  gemm_qkv<_Float16, 2><<<gp, 256, 0, stream>>>(Oi, wo, bo, d_out);
}
